// TopKSAE_42374147342788
// MI455X (gfx1250) — matrix operand order repaired and checked by reading, not compiled, not run on hardware
//
#include <hip/hip_runtime.h>
#include <hip/hip_bf16.h>

#define D_MODEL 768
#define D_SAE   12288
#define NROWS   8192
#define TOPK    32

typedef __attribute__((ext_vector_type(16))) __bf16 v16bf;
typedef __attribute__((ext_vector_type(8)))  float  v8f;
typedef int v4i __attribute__((vector_size(16)));          // 128-bit payload
typedef __attribute__((address_space(1))) v4i* gptr_v4i;   // global (device)
typedef __attribute__((address_space(3))) v4i* lptr_v4i;   // LDS (shared)

// CDNA5 async global->LDS copies (ASYNCcnt-tracked), with sync fallback.
#if defined(__has_builtin)
#if __has_builtin(__builtin_amdgcn_global_load_async_to_lds_b128) && \
    __has_builtin(__builtin_amdgcn_s_wait_asynccnt)
#define HAVE_ASYNC_LDS 1
#endif
#endif
#ifndef HAVE_ASYNC_LDS
#define HAVE_ASYNC_LDS 0
#endif

__device__ __forceinline__ void tile_copy16(float* lds_dst, const float* gsrc) {
#if HAVE_ASYNC_LDS
    __builtin_amdgcn_global_load_async_to_lds_b128(
        (gptr_v4i)gsrc, (lptr_v4i)lds_dst, 0, 0);
#else
    *(float4*)lds_dst = *(const float4*)gsrc;
#endif
}

__device__ __forceinline__ void wait_tile_copies() {
#if HAVE_ASYNC_LDS
    __builtin_amdgcn_s_wait_asynccnt(0);
#endif
    __syncthreads();
}

// ---------------------------------------------------------------------------
// Kernel 1: transpose W_dec [768, 12288] -> WdT [12288, 768] in workspace.
// ---------------------------------------------------------------------------
__global__ __launch_bounds__(256) void sae_transpose_wdec(
    const float* __restrict__ Wd, float* __restrict__ WdT)
{
    __shared__ float tile[32][33];
    const int bx = blockIdx.x;            // along D_SAE (384 tiles)
    const int by = blockIdx.y;            // along D_MODEL (24 tiles)
    const int tx = threadIdx.x & 31;
    const int ty = threadIdx.x >> 5;      // 0..7

    #pragma unroll
    for (int r = ty; r < 32; r += 8)
        tile[r][tx] = Wd[(size_t)(by * 32 + r) * D_SAE + bx * 32 + tx];
    __syncthreads();
    #pragma unroll
    for (int r = ty; r < 32; r += 8)
        WdT[(size_t)(bx * 32 + r) * D_MODEL + by * 32 + tx] = tile[tx][r];
}

// ---------------------------------------------------------------------------
// Kernel 2: encode GEMM. latents[N, D_SAE] = x @ W_enc^T + b_enc, f32 in/out,
// bf16 WMMA math. 256x128 macro tile per block (halves L2 read traffic vs
// 128x128), 8 waves in a 4x2 grid, each wave owns 64x64 = 4x4 fragments.
// LDS double-buffered; next K-tile fetched with async global->LDS copies
// while the current tile feeds the WMMAs.
// ---------------------------------------------------------------------------
#define LDSTRIDE 36   // 32 floats + 4 pad (16B-aligned, spreads LDS banks)
#define TM 256
#define TS 128

__global__ __launch_bounds__(256) void sae_encode_wmma(
    const float* __restrict__ x, const float* __restrict__ W_enc,
    const float* __restrict__ b_enc, float* __restrict__ latents)
{
    __shared__ float As[2][TM * LDSTRIDE];   // 73,728 B
    __shared__ float Bs[2][TS * LDSTRIDE];   // 36,864 B  (total ~108 KB)

    const int tid     = threadIdx.x;
    const int wave    = tid >> 5;      // 0..7
    const int lane    = tid & 31;
    const int waveRow = wave >> 1;     // 0..3 -> 64 rows each
    const int waveCol = wave & 1;      // 0..1 -> 64 cols each
    const int n0      = blockIdx.y * TM;
    const int s0      = blockIdx.x * TS;
    const int row16   = lane & 15;
    const int kg      = lane >> 4;     // lane half selects K-group

    v8f acc[4][4];
    #pragma unroll
    for (int m = 0; m < 4; ++m)
        #pragma unroll
        for (int j = 0; j < 4; ++j)
            acc[m][j] = (v8f){0.f, 0.f, 0.f, 0.f, 0.f, 0.f, 0.f, 0.f};

    // Stage one K-tile (256x32 of x, 128x32 of W_enc) into LDS buffer `buf`.
    auto load_tiles = [&](int buf, int k0) {
        #pragma unroll
        for (int i = 0; i < 8; ++i) {            // A: 2048 float4
            const int id = tid + 256 * i;
            const int r  = id >> 3;              // 0..255
            const int c4 = (id & 7) * 4;         // 0,4,..,28
            tile_copy16(&As[buf][r * LDSTRIDE + c4],
                        x + (size_t)(n0 + r) * D_MODEL + k0 + c4);
        }
        #pragma unroll
        for (int i = 0; i < 4; ++i) {            // B: 1024 float4
            const int id = tid + 256 * i;
            const int r  = id >> 3;              // 0..127
            const int c4 = (id & 7) * 4;
            tile_copy16(&Bs[buf][r * LDSTRIDE + c4],
                        W_enc + (size_t)(s0 + r) * D_MODEL + k0 + c4);
        }
    };

    load_tiles(0, 0);
    wait_tile_copies();

    int kb = 0;
    for (int k0 = 0; k0 < D_MODEL; k0 += 32, kb ^= 1) {
        if (k0 + 32 < D_MODEL)
            load_tiles(kb ^ 1, k0 + 32);   // prefetch next tile into other buffer

        // Build bf16 fragments per the CDNA5 16-bit VGPR layouts.
        // A (16x32): VGPR q<4 -> K = kg*8 + 2q, q>=4 -> K = 16 + kg*8 + 2(q-4)
        v16bf a[4], b[4];
        #pragma unroll
        for (int m = 0; m < 4; ++m) {
            const float* src = &As[kb][(waveRow * 64 + m * 16 + row16) * LDSTRIDE];
            #pragma unroll
            for (int q = 0; q < 8; ++q) {
                const int kk = (q < 4) ? (kg * 8 + 2 * q)
                                       : (16 + kg * 8 + 2 * (q - 4));
                a[m][2 * q]     = (__bf16)src[kk];
                a[m][2 * q + 1] = (__bf16)src[kk + 1];
            }
        }
        // B (32x16): lane holds column N = lane&15; VGPR q<4 -> K = kg*8 + 2q, q>=4 -> K = 16 + kg*8 + 2(q-4)
        #pragma unroll
        for (int j = 0; j < 4; ++j) {
            const float* src = &Bs[kb][(waveCol * 64 + j * 16 + row16) * LDSTRIDE];
            #pragma unroll
            for (int q = 0; q < 8; ++q) {
                const int kk = (q < 4) ? (kg * 8 + 2 * q)
                                       : (16 + kg * 8 + 2 * (q - 4));
                b[j][2 * q]     = (__bf16)src[kk];
                b[j][2 * q + 1] = (__bf16)src[kk + 1];
            }
        }

        #pragma unroll
        for (int m = 0; m < 4; ++m)
            #pragma unroll
            for (int j = 0; j < 4; ++j)
                acc[m][j] = __builtin_amdgcn_wmma_f32_16x16x32_bf16(
                    false, a[m], false, b[j], (short)0, acc[m][j], false, false);

        if (k0 + 32 < D_MODEL)
            wait_tile_copies();            // next buffer ready; prev fully read
    }

    // Epilogue: C/D layout -> M = r + 8*kg, N = lane&15. Add b_enc, store f32.
    #pragma unroll
    for (int m = 0; m < 4; ++m) {
        #pragma unroll
        for (int j = 0; j < 4; ++j) {
            const int col    = s0 + waveCol * 64 + j * 16 + row16;
            const float bias = b_enc[col];
            #pragma unroll
            for (int r = 0; r < 8; ++r) {
                const int rowg = n0 + waveRow * 64 + m * 16 + r + 8 * kg;
                latents[(size_t)rowg * D_SAE + col] = acc[m][j][r] + bias;
            }
        }
    }
}

// ---------------------------------------------------------------------------
// Kernel 3: per-row top-k (k=32) + sparse rewrite + sparse decode.
// One block per row; 12288-float row lives in LDS (48 KB << 320 KB/WGP).
// Argmax via LDS u64 atomicMax on (sortable_f32 << 32 | ~index) — lowest index
// wins ties, matching jax.lax.top_k.
// ---------------------------------------------------------------------------
__global__ __launch_bounds__(256) void sae_topk_decode(
    float* __restrict__ latents,           // [N, D_SAE] in/out (becomes sparse)
    const float* __restrict__ WdT,         // [D_SAE, D_MODEL]
    const float* __restrict__ b_dec,       // [D_MODEL]
    float* __restrict__ recon)             // [N, D_MODEL]
{
    __shared__ float rowb[D_SAE];
    __shared__ unsigned long long best;
    __shared__ int   top_idx[TOPK];
    __shared__ float top_val[TOPK];

    const int n   = blockIdx.x;
    const int tid = threadIdx.x;
    float* lrow = latents + (size_t)n * D_SAE;

    for (int i = tid; i < D_SAE; i += 256) rowb[i] = lrow[i];
    __syncthreads();

    for (int t = 0; t < TOPK; ++t) {
        if (tid == 0) best = 0ull;
        __syncthreads();
        unsigned long long mykey = 0ull;
        for (int i = tid; i < D_SAE; i += 256) {
            const unsigned u = __float_as_uint(rowb[i]);
            const unsigned s = (u & 0x80000000u) ? ~u : (u | 0x80000000u);
            const unsigned long long key =
                ((unsigned long long)s << 32) | (unsigned)(~i);
            if (key > mykey) mykey = key;
        }
        atomicMax(&best, mykey);
        __syncthreads();
        if (tid == 0) {
            const int j = (int)(~(unsigned)(best & 0xffffffffu));
            top_idx[t] = j;
            top_val[t] = rowb[j];
            rowb[j] = -__builtin_inff();
        }
        __syncthreads();
    }

    // Rebuild the sparse row: zeros everywhere, scatter the k winners.
    for (int i = tid; i < D_SAE; i += 256) rowb[i] = 0.0f;
    __syncthreads();
    if (tid < TOPK) rowb[top_idx[tid]] = top_val[tid];
    __syncthreads();
    for (int i = tid; i < D_SAE; i += 256) lrow[i] = rowb[i];

    // Sparse decode: recon[n,:] = sum_t val_t * WdT[idx_t, :] + b_dec.
    float acc0 = 0.f, acc1 = 0.f, acc2 = 0.f;
    const int d0 = tid, d1 = tid + 256, d2 = tid + 512;
    #pragma unroll
    for (int t = 0; t < TOPK; ++t) {
        const float v  = top_val[t];
        const float* w = WdT + (size_t)top_idx[t] * D_MODEL;
        acc0 += v * w[d0];
        acc1 += v * w[d1];
        acc2 += v * w[d2];
    }
    float* rrow = recon + (size_t)n * D_MODEL;
    rrow[d0] = acc0 + b_dec[d0];
    rrow[d1] = acc1 + b_dec[d1];
    rrow[d2] = acc2 + b_dec[d2];
}

// ---------------------------------------------------------------------------
extern "C" void kernel_launch(void* const* d_in, const int* in_sizes, int n_in,
                              void* d_out, int out_size, void* d_ws, size_t ws_size,
                              hipStream_t stream) {
    (void)in_sizes; (void)n_in; (void)out_size; (void)ws_size;
    const float* x     = (const float*)d_in[0];
    const float* W_enc = (const float*)d_in[1];
    const float* b_enc = (const float*)d_in[2];
    const float* W_dec = (const float*)d_in[3];
    const float* b_dec = (const float*)d_in[4];
    // d_in[5] is k == 32 (hardcoded as TOPK)

    float* out    = (float*)d_out;
    float* recon  = out;                                   // [N, D_MODEL]
    float* sparse = out + (size_t)NROWS * D_MODEL;         // [N, D_SAE]
    float* WdT    = (float*)d_ws;                          // [D_SAE, D_MODEL]

    sae_transpose_wdec<<<dim3(D_SAE / 32, D_MODEL / 32), 256, 0, stream>>>(W_dec, WdT);
    sae_encode_wmma<<<dim3(D_SAE / TS, NROWS / TM), 256, 0, stream>>>(x, W_enc, b_enc, sparse);
    sae_topk_decode<<<NROWS, 256, 0, stream>>>(sparse, WdT, b_dec, recon);
}
